// RWKV_24756191494370
// MI455X (gfx1250) — matrix operand order repaired and checked by reading, not compiled, not run on hardware
//
#include <hip/hip_runtime.h>

// ---------------------------------------------------------------------------
// RWKV forward for MI455X (gfx1250, wave32, WMMA).
// GEMMs: x @ W.T with K-major f32 operands, converted f32->bf16 in-register,
// staged through double-buffered LDS (block tile 256x128, 8 waves, each wave
// a 64x64 register tile -> 16 v_wmma_f32_16x16x32_bf16 per K-step). The
// workload is cache-traffic bound (~1.3 TFLOP vs ~600MB of weights), so the
// design maximizes reuse: registers (4x4 frag product), LDS (A shared by 2
// waves, W by 4), L2 (per-layer weights << 192MB).
// WKV: chunk-parallel 3-pass scan (log-scaled state is a linear recurrence:
// decay = pp += L*w, merge = max-shift add).
// ---------------------------------------------------------------------------

typedef __attribute__((ext_vector_type(16))) __bf16 bf16x16;
typedef __attribute__((ext_vector_type(8)))  __bf16 bf16x8;
typedef __attribute__((ext_vector_type(8)))  float  f32x8;
typedef __attribute__((ext_vector_type(4)))  float  f32x4;

__device__ __forceinline__ bf16x8 cvt8(f32x4 a, f32x4 b) {
  bf16x8 r;
#pragma unroll
  for (int i = 0; i < 4; ++i) { r[i] = (__bf16)a[i]; r[i + 4] = (__bf16)b[i]; }
  return r;
}
__device__ __forceinline__ bf16x16 join8(bf16x8 a, bf16x8 b) {
  return __builtin_shufflevector(a, b, 0, 1, 2, 3, 4, 5, 6, 7,
                                 8, 9, 10, 11, 12, 13, 14, 15);
}

#define BM   256            // block tile M (4 waves of 64)
#define BN   128            // block tile N (2 waves of 64)
#define KB   32             // K step
#define LROW 40             // LDS row stride in halves (32 + 8 pad: bank-safe)

// C[M,N] = A[M,K] @ W[N,K]^T (+ optional bias[N]).
// A, W row-major f32 (K contiguous). Double-buffered LDS staging; fragment
// layouts per CDNA5 ISA 7.12.2 (16-bit A 16x32, B 32x16, f32 C/D).
__global__ __launch_bounds__(256) void wmma_gemm_kernel(
    const float* __restrict__ A, const float* __restrict__ W,
    const float* __restrict__ bias, float* __restrict__ C,
    int M, int N, int K)
{
  __shared__ __bf16 lA[2][BM * LROW];   // 40 KB
  __shared__ __bf16 lB[2][BN * LROW];   // 20 KB

  const int tid   = threadIdx.x;
  const int wave  = tid >> 5;
  const int lane  = tid & 31;
  const int waveM = wave >> 1;                   // 0..3
  const int waveN = wave & 1;                    // 0..1
  const int blocksN = N / BN;
  const int bm = blockIdx.x / blocksN;
  const int bn = blockIdx.x % blocksN;

  // cooperative staging: thread t owns A row t (32 floats) and half of B row t/2
  const int arow  = tid;                         // 0..255
  const int brow  = tid >> 1;                    // 0..127
  const int bkoff = (tid & 1) * 16;              // 0 or 16
  const float* Ag = A + (size_t)(bm * BM + arow) * K;
  const float* Wg = W + (size_t)(bn * BN + brow) * K + bkoff;

  // fragment read positions (ISA VGPR layouts)
  const int l15 = lane & 15;
  const int hi  = (lane >= 16);
  const int k0a = hi ? 8 : 0;                    // A-frag K base
  const int k0b = hi ? 8 : 0;                    // B-frag K base
  const int aBase = (waveM * 64 + l15) * LROW + k0a;
  const int bBase = (waveN * 64 + l15) * LROW + k0b;

  f32x8 acc[4][4] = {};                          // acc[i][j]: rows 16i, cols 16j

  f32x4 pa[8];
  f32x4 pb[4];
#pragma unroll
  for (int i = 0; i < 8; ++i) pa[i] = *(const f32x4*)(Ag + 4 * i);
#pragma unroll
  for (int i = 0; i < 4; ++i) pb[i] = *(const f32x4*)(Wg + 4 * i);
  {
    __bf16* dA = &lA[0][arow * LROW];
#pragma unroll
    for (int i = 0; i < 4; ++i)
      *(bf16x8*)(dA + 8 * i) = cvt8(pa[2 * i], pa[2 * i + 1]);
    __bf16* dB = &lB[0][brow * LROW + bkoff];
#pragma unroll
    for (int i = 0; i < 2; ++i)
      *(bf16x8*)(dB + 8 * i) = cvt8(pb[2 * i], pb[2 * i + 1]);
  }

  for (int kb = 0; kb < K; kb += KB) {
    const int  cur  = (kb >> 5) & 1;
    const bool more = (kb + KB) < K;
    if (more) {                                  // issue next tile's loads early
#pragma unroll
      for (int i = 0; i < 8; ++i) pa[i] = *(const f32x4*)(Ag + kb + KB + 4 * i);
#pragma unroll
      for (int i = 0; i < 4; ++i) pb[i] = *(const f32x4*)(Wg + kb + KB + 4 * i);
    }
    __syncthreads();                             // lds[cur] visible to all waves

    bf16x16 af[4];
#pragma unroll
    for (int i = 0; i < 4; ++i) {
      const __bf16* p = &lA[cur][aBase + 16 * i * LROW];
      af[i] = join8(*(const bf16x8*)(p), *(const bf16x8*)(p + 16));
    }
#pragma unroll
    for (int j = 0; j < 4; ++j) {
      const __bf16* q = &lB[cur][bBase + 16 * j * LROW];
      bf16x16 bf = join8(*(const bf16x8*)(q), *(const bf16x8*)(q + 16));
#pragma unroll
      for (int i = 0; i < 4; ++i)
        acc[i][j] = __builtin_amdgcn_wmma_f32_16x16x32_bf16(
            false, af[i], false, bf, (short)0, acc[i][j], false, false);
    }

    if (more) {                                  // fill the other buffer
      __bf16* dA = &lA[cur ^ 1][arow * LROW];
#pragma unroll
      for (int i = 0; i < 4; ++i)
        *(bf16x8*)(dA + 8 * i) = cvt8(pa[2 * i], pa[2 * i + 1]);
      __bf16* dB = &lB[cur ^ 1][brow * LROW + bkoff];
#pragma unroll
      for (int i = 0; i < 2; ++i)
        *(bf16x8*)(dB + 8 * i) = cvt8(pb[2 * i], pb[2 * i + 1]);
    }
  }

  // C/D layout: VGPR r -> row (16i + r + (lane<16 ? 0 : 8)), col = lane&15.
  float bv[4];
#pragma unroll
  for (int j = 0; j < 4; ++j)
    bv[j] = bias ? bias[bn * BN + waveN * 64 + 16 * j + l15] : 0.f;
  const int rOut = bm * BM + waveM * 64 + (hi ? 8 : 0);
  const int cOut = bn * BN + waveN * 64 + l15;
#pragma unroll
  for (int i = 0; i < 4; ++i) {
    float* Cp = C + (size_t)(rOut + 16 * i) * N + cOut;
#pragma unroll
    for (int j = 0; j < 4; ++j)
#pragma unroll
      for (int r = 0; r < 8; ++r)
        Cp[(size_t)r * N + 16 * j] = acc[i][j][r] + bv[j];
  }
}

// LayerNorm over E per token. One block (256 thr, 8 waves) per token.
// Safe in-place: each thread caches its own elements in registers first.
__global__ __launch_bounds__(256) void layernorm_kernel(
    const float* __restrict__ x, const float* __restrict__ g,
    const float* __restrict__ b, float* __restrict__ out, int E)
{
  __shared__ float sm1[8], sm2[8];
  const int tid = threadIdx.x;
  const float* xr = x + (size_t)blockIdx.x * E;
  float* orow = out + (size_t)blockIdx.x * E;
  const int iters = E >> 8;          // E multiple of 256, iters <= 8
  float v[8];
  float s1 = 0.f, s2 = 0.f;
  for (int i = 0; i < iters; ++i) {
    float t = xr[tid + (i << 8)];
    v[i] = t; s1 += t; s2 += t * t;
  }
#pragma unroll
  for (int o = 16; o > 0; o >>= 1) {
    s1 += __shfl_xor(s1, o, 32);
    s2 += __shfl_xor(s2, o, 32);
  }
  if ((tid & 31) == 0) { sm1[tid >> 5] = s1; sm2[tid >> 5] = s2; }
  __syncthreads();
  float t1 = 0.f, t2 = 0.f;
#pragma unroll
  for (int i = 0; i < 8; ++i) { t1 += sm1[i]; t2 += sm2[i]; }
  const float mean = t1 / (float)E;
  const float var  = t2 / (float)E - mean * mean;
  const float rstd = rsqrtf(var + 1e-5f);
  for (int i = 0; i < iters; ++i) {
    int idx = tid + (i << 8);
    orow[idx] = (v[i] - mean) * rstd * g[idx] + b[idx];
  }
}

// ---- WKV, chunk-parallel 3-pass scan --------------------------------------
// State (aa,bb,pp) represents e^pp * (aa,bb); per-step update:
//   pp' = max(pp+w, kt); aa' = e^(pp+w-pp')aa + e^(kt-pp')vt; bb likewise w/ 1.
// Decaying L steps with no input is pp += L*w; merging two states is a
// max-shifted add. pass1 = per-chunk local state from identity,
// pass2 = serial prefix over the 16 chunk states per channel,
// pass3 = replay each chunk from its carry-in, emitting gated y.

__global__ __launch_bounds__(256) void wkv_chunk_state_kernel(
    const float* __restrict__ decay, const float* __restrict__ k,
    const float* __restrict__ r,
    float* __restrict__ Saa, float* __restrict__ Sbb, float* __restrict__ Spp,
    int B, int T, int E, int chunk)
{
  const int BE = B * E;
  const int nc = T / chunk;
  const int idx = blockIdx.x * blockDim.x + threadIdx.x;   // c*BE + be
  if (idx >= BE * nc) return;
  const int c  = idx / BE;
  const int be = idx - c * BE;
  const int b  = be / E, e = be - b * E;
  const float w = -__expf(decay[e]);
  float aa = 0.f, bb = 0.f, pp = -1e38f;
  size_t off = ((size_t)b * T + (size_t)c * chunk) * E + e;
  for (int t = 0; t < chunk; ++t) {
    const float kt = k[off], vt = r[off];
    float w2 = pp + w;
    float p2 = fmaxf(w2, kt);
    float e1 = __expf(w2 - p2);
    float e2 = __expf(kt - p2);
    aa = e1 * aa + e2 * vt;
    bb = e1 * bb + e2;
    pp = p2;
    off += (size_t)E;
  }
  Saa[idx] = aa; Sbb[idx] = bb; Spp[idx] = pp;
}

__global__ __launch_bounds__(256) void wkv_prefix_kernel(
    const float* __restrict__ decay,
    const float* __restrict__ Saa, const float* __restrict__ Sbb,
    const float* __restrict__ Spp,
    float* __restrict__ Caa, float* __restrict__ Cbb, float* __restrict__ Cpp,
    int B, int T, int E, int chunk)
{
  const int BE = B * E;
  const int be = blockIdx.x * blockDim.x + threadIdx.x;
  if (be >= BE) return;
  const int e = be % E;
  const float w   = -__expf(decay[e]);
  const float shw = (float)chunk * w;          // chunk-length decay in log space
  const int nc = T / chunk;
  float aa = 0.f, bb = 0.f, pp = -1e38f;
  for (int c = 0; c < nc; ++c) {
    const int i = c * BE + be;
    Caa[i] = aa; Cbb[i] = bb; Cpp[i] = pp;     // carry-in for chunk c
    const float sh = pp + shw;
    const float lp = Spp[i];
    const float p  = fmaxf(sh, lp);
    const float ec = __expf(sh - p);
    const float el = __expf(lp - p);
    aa = ec * aa + el * Saa[i];
    bb = ec * bb + el * Sbb[i];
    pp = p;
  }
}

__global__ __launch_bounds__(256) void wkv_emit_kernel(
    const float* __restrict__ decay, const float* __restrict__ first,
    const float* __restrict__ k, const float* __restrict__ r,
    const float* __restrict__ Caa, const float* __restrict__ Cbb,
    const float* __restrict__ Cpp,
    float* __restrict__ y, int B, int T, int E, int chunk)
{
  const int BE = B * E;
  const int nc = T / chunk;
  const int idx = blockIdx.x * blockDim.x + threadIdx.x;   // c*BE + be
  if (idx >= BE * nc) return;
  const int c  = idx / BE;
  const int be = idx - c * BE;
  const int b  = be / E, e = be - b * E;
  const float w = -__expf(decay[e]);
  const float u = first[e];
  float aa = Caa[idx], bb = Cbb[idx], pp = Cpp[idx];
  size_t off = ((size_t)b * T + (size_t)c * chunk) * E + e;
  for (int t = 0; t < chunk; ++t) {
    const float kt = k[off];
    const float vt = r[off];                   // v == r in this model
    float ww = u + kt;
    float p  = fmaxf(pp, ww);
    float e1 = __expf(pp - p);
    float e2 = __expf(ww - p);
    float yv = (e1 * aa + e2 * vt) / (e1 * bb + e2);
    y[off] = yv / (1.f + __expf(-vt));         // fused sigmoid(r) gate
    float w2 = pp + w;
    float p2 = fmaxf(w2, kt);
    float eb1 = __expf(w2 - p2);
    float eb2 = __expf(kt - p2);
    aa = eb1 * aa + eb2 * vt;
    bb = eb1 * bb + eb2;
    pp = p2;
    off += (size_t)E;
  }
}

// ---- elementwise epilogues ------------------------------------------------

__global__ void relusq_kernel(float* __restrict__ h, long n) {
  long i = (long)blockIdx.x * blockDim.x + threadIdx.x;
  if (i < n) { float t = fmaxf(h[i], 0.f); h[i] = t * t; }
}

__global__ void sigmul_add_kernel(float* __restrict__ res,
                                  const float* __restrict__ rg,
                                  const float* __restrict__ hv, long n) {
  long i = (long)blockIdx.x * blockDim.x + threadIdx.x;
  if (i < n) res[i] += hv[i] / (1.f + __expf(-rg[i]));
}

__global__ void add_kernel(float* __restrict__ res, const float* __restrict__ s, long n) {
  long i = (long)blockIdx.x * blockDim.x + threadIdx.x;
  if (i < n) res[i] += s[i];
}

// ---------------------------------------------------------------------------

extern "C" void kernel_launch(void* const* d_in, const int* in_sizes, int n_in,
                              void* d_out, int out_size, void* d_ws, size_t ws_size,
                              hipStream_t stream) {
  (void)in_sizes; (void)n_in; (void)out_size; (void)ws_size;
  const int B = 4, T = 1024, E = 1024, D = 512, Lm1 = 11;
  const int BT = B * T;
  const int E4 = 4 * E;
  const int CHUNK = 64;
  const int NC = T / CHUNK;
  const size_t BTE = (size_t)BT * E;
  const size_t BE  = (size_t)B * E;

  const float* inputs   = (const float*)d_in[0];
  const float* W_in     = (const float*)d_in[1];
  const float* b_in     = (const float*)d_in[2];
  const float* l0_ln0_g = (const float*)d_in[3];
  const float* l0_ln0_b = (const float*)d_in[4];
  const float* l0_ln1_g = (const float*)d_in[5];
  const float* l0_ln1_b = (const float*)d_in[6];
  const float* l0_ln2_g = (const float*)d_in[7];
  const float* l0_ln2_b = (const float*)d_in[8];
  const float* l0_cm_k  = (const float*)d_in[9];
  const float* l0_cm_v  = (const float*)d_in[10];
  const float* l0_cm_r  = (const float*)d_in[11];
  const float* l0_ff_k  = (const float*)d_in[12];
  const float* l0_ff_v  = (const float*)d_in[13];
  const float* l0_ff_r  = (const float*)d_in[14];
  const float* tm_k     = (const float*)d_in[15];
  const float* tm_r     = (const float*)d_in[16];
  const float* tm_o     = (const float*)d_in[17];
  const float* tm_decay = (const float*)d_in[18];
  const float* tm_first = (const float*)d_in[19];
  const float* ln1_g    = (const float*)d_in[20];
  const float* ln1_b    = (const float*)d_in[21];
  const float* ln2_g    = (const float*)d_in[22];
  const float* ln2_b    = (const float*)d_in[23];
  const float* ff_k     = (const float*)d_in[24];
  const float* ff_v     = (const float*)d_in[25];
  const float* ff_r     = (const float*)d_in[26];
  const float* out_g    = (const float*)d_in[27];
  const float* out_b    = (const float*)d_in[28];

  // Workspace: 5*[BT,E] + [BT,4E] f32 (144 MB) + 6*[NC,B*E] wkv states (1.5 MB)
  float* res = (float*)d_ws;
  float* xn  = res + BTE;
  float* kb  = xn  + BTE;
  float* rb  = kb  + BTE;
  float* yb  = rb  + BTE;
  float* h4  = yb  + BTE;
  float* Saa = h4  + (size_t)BT * E4;
  float* Sbb = Saa + BE * NC;
  float* Spp = Sbb + BE * NC;
  float* Caa = Spp + BE * NC;
  float* Cbb = Caa + BE * NC;
  float* Cpp = Cbb + BE * NC;

  auto gemm = [&](const float* A, const float* W, const float* bias, float* C,
                  int M, int N, int K) {
    int blocks = (M / BM) * (N / BN);
    wmma_gemm_kernel<<<blocks, 256, 0, stream>>>(A, W, bias, C, M, N, K);
  };
  auto ln = [&](const float* x, const float* g, const float* b, float* o) {
    layernorm_kernel<<<BT, 256, 0, stream>>>(x, g, b, o, E);
  };
  auto ewb = [](long n) { return (unsigned)((n + 255) / 256); };

  auto wkv = [&](const float* dec, const float* fir) {
    const int nCS = (int)BE * NC;
    wkv_chunk_state_kernel<<<(nCS + 255) / 256, 256, 0, stream>>>(
        dec, kb, rb, Saa, Sbb, Spp, B, T, E, CHUNK);
    wkv_prefix_kernel<<<((int)BE + 255) / 256, 256, 0, stream>>>(
        dec, Saa, Sbb, Spp, Caa, Cbb, Cpp, B, T, E, CHUNK);
    wkv_emit_kernel<<<(nCS + 255) / 256, 256, 0, stream>>>(
        dec, fir, kb, rb, Caa, Cbb, Cpp, yb, B, T, E, CHUNK);
  };

  // channel_mix: res += sigmoid(LN@Wr.T) * (relu(LN@Wk.T)^2 @ Wv.T)
  auto channel_mix = [&](const float* lng, const float* lnb,
                         const float* Wk, const float* Wv, const float* Wr) {
    ln(res, lng, lnb, xn);
    gemm(xn, Wk, nullptr, h4, BT, E4, E);
    relusq_kernel<<<ewb((long)BT * E4), 256, 0, stream>>>(h4, (long)BT * E4);
    gemm(h4, Wv, nullptr, rb, BT, E, E4);
    gemm(xn, Wr, nullptr, kb, BT, E, E);
    sigmul_add_kernel<<<ewb((long)BTE), 256, 0, stream>>>(res, kb, rb, (long)BTE);
  };

  // ---- embed: res = inputs @ W_in.T + b_in ----
  gemm(inputs, W_in, b_in, res, BT, E, D);

  // ---- layer 0: ln0, then two channel-mixes ----
  ln(res, l0_ln0_g, l0_ln0_b, res);
  channel_mix(l0_ln1_g, l0_ln1_b, l0_cm_k, l0_cm_v, l0_cm_r);
  channel_mix(l0_ln2_g, l0_ln2_b, l0_ff_k, l0_ff_v, l0_ff_r);

  // ---- layers 1..11: TimeMix + ChannelMix ----
  for (int l = 0; l < Lm1; ++l) {
    const size_t EE = (size_t)E * E, E4E = (size_t)E4 * E;
    ln(res, ln1_g + (size_t)l * E, ln1_b + (size_t)l * E, xn);
    gemm(xn, tm_k + l * EE, nullptr, kb, BT, E, E);
    gemm(xn, tm_r + l * EE, nullptr, rb, BT, E, E);
    wkv(tm_decay + (size_t)l * E, tm_first + (size_t)l * E);
    gemm(yb, tm_o + l * EE, nullptr, kb, BT, E, E);
    add_kernel<<<ewb((long)BTE), 256, 0, stream>>>(res, kb, (long)BTE);
    ln(res, ln2_g + (size_t)l * E, ln2_b + (size_t)l * E, xn);
    gemm(xn, ff_k + l * E4E, nullptr, h4, BT, E4, E);
    relusq_kernel<<<ewb((long)BT * E4), 256, 0, stream>>>(h4, (long)BT * E4);
    gemm(h4, ff_v + l * E4E, nullptr, rb, BT, E, E4);
    gemm(xn, ff_r + l * EE, nullptr, kb, BT, E, E);
    sigmul_add_kernel<<<ewb((long)BTE), 256, 0, stream>>>(res, kb, rb, (long)BTE);
  }

  // ---- final LN -> d_out ----
  ln(res, out_g, out_b, (float*)d_out);
}
